// CustomBertModel_32521492365998
// MI455X (gfx1250) — compile-verified, matrix operand order checked by reading, not run on hardware
//
#include <hip/hip_runtime.h>
#include <math.h>

// ---------------- model constants ----------------
#define BB   16
#define SS   512
#define HH   768
#define BS   (BB * SS)     // 8192 rows
#define HD   64
#define EPS  1e-12f

typedef _Float16 half_t;
typedef __attribute__((ext_vector_type(8)))  _Float16 v8h;
typedef __attribute__((ext_vector_type(16))) _Float16 v16h;
typedef __attribute__((ext_vector_type(8)))  float    v8f;

// D = A(16x32 f16) x B(32x16 f16) + C (f32), CDNA5 wave32 WMMA
static __device__ __forceinline__ v8f wmma16(v16h a, v16h b, v8f c) {
  return __builtin_amdgcn_wmma_f32_16x16x32_f16(
      /*neg_a=*/false, a, /*neg_b=*/false, b,
      /*c_mod=*/(short)0, c, /*reuse_a=*/false, /*reuse_b=*/false);
}

// 16x32 fragment (A layout; also B^T layout) from f16 row-major memory/LDS.
// ISA: lane L holds row M=L&15; element e -> K = (e&7) + (e>>3)*16 + (L>>4)*8.
// Per lane this is two contiguous 8-half (16B) runs -> two b128 loads.
static __device__ __forceinline__ v16h load_frag16(const half_t* p, int ld, int lane) {
  const half_t* row = p + (size_t)(lane & 15) * ld + ((lane >> 4) << 3);
  v8h lo = *(const v8h*)(row);        // K = kOff + 0..7
  v8h hi = *(const v8h*)(row + 16);   // K = 16 + kOff + 0..7
  return __builtin_shufflevector(lo, hi, 0, 1, 2, 3, 4, 5, 6, 7,
                                         8, 9, 10, 11, 12, 13, 14, 15);
}

// ---------------- weight transpose + f32->f16 convert ----------------
// in: [batch][K][N] f32 (batch = blockIdx.y) -> out: [batch][N][K] f16
__global__ void wt_cvt(const float* __restrict__ in, half_t* __restrict__ out,
                       int K, int N) {
  size_t per = (size_t)K * N;
  size_t idx = (size_t)blockIdx.x * 256 + threadIdx.x;
  if (idx >= per) return;
  size_t base = (size_t)blockIdx.y * per;
  int k = (int)(idx / N), n = (int)(idx % N);
  out[base + (size_t)n * K + k] = (half_t)in[base + idx];
}

// ---------------- embeddings: o_m = LN(embed_m + pos) -> f16 ----------------
__global__ __launch_bounds__(128) void embed_ln(
    const float* __restrict__ e1, const float* __restrict__ e2,
    const float* __restrict__ e3, const float* __restrict__ pos,
    const float* __restrict__ w, const float* __restrict__ b,
    half_t* __restrict__ o) {
  int gid  = blockIdx.x * 4 + (threadIdx.x >> 5);
  int lane = threadIdx.x & 31;
  if (gid >= 3 * BS) return;
  int m = gid / BS, row = gid % BS, s = row & (SS - 1);
  const float* e = (m == 0) ? e1 : (m == 1) ? e2 : e3;
  const float* er = e + (size_t)row * HH;
  const float* prow = pos + (size_t)s * HH;
  float vals[24];
  float sum = 0.f;
#pragma unroll
  for (int i = 0; i < 24; ++i) {
    int c = lane + i * 32;
    vals[i] = er[c] + prow[c];
    sum += vals[i];
  }
#pragma unroll
  for (int off = 16; off >= 1; off >>= 1) sum += __shfl_xor(sum, off, 32);
  float mean = sum * (1.f / HH);
  float var = 0.f;
#pragma unroll
  for (int i = 0; i < 24; ++i) { float d = vals[i] - mean; var += d * d; }
#pragma unroll
  for (int off = 16; off >= 1; off >>= 1) var += __shfl_xor(var, off, 32);
  float inv = rsqrtf(var * (1.f / HH) + EPS);
  half_t* orow = o + (size_t)gid * HH;
#pragma unroll
  for (int i = 0; i < 24; ++i) {
    int c = lane + i * 32;
    orow[c] = (half_t)((vals[i] - mean) * inv * w[c] + b[c]);
  }
}

// res0 = (o0 + o1 + o2) / 3  (f16 in, f16 out)
__global__ void res_avg(const half_t* __restrict__ o, half_t* __restrict__ r) {
  size_t i = (size_t)blockIdx.x * blockDim.x + threadIdx.x;
  size_t N = (size_t)BS * HH;
  if (i < N)
    r[i] = (half_t)(((float)o[i] + (float)o[i + N] + (float)o[i + 2 * N]) * (1.f / 3.f));
}

// ---------------- QKV projection GEMM ----------------
// grid = (BS/64, 12 = m*4+h, 3 = q/k/v). Q,K: [(m*16+b)*4+h][S][64] f16.
// V written TRANSPOSED: [(m*16+b)*4+h][64][S] f16 (for fast probs x V frags).
__global__ __launch_bounds__(128) void qkv_gemm(
    const half_t* __restrict__ X, long modStride, int lda, int din,
    const half_t* __restrict__ Wqt, const float* __restrict__ bq,
    const half_t* __restrict__ Wkt, const float* __restrict__ bk,
    const half_t* __restrict__ Wvt, const float* __restrict__ bv,
    half_t* __restrict__ Q, half_t* __restrict__ K, half_t* __restrict__ V) {
  int wave = threadIdx.x >> 5, lane = threadIdx.x & 31;
  int mh = blockIdx.y, m = mh >> 2, h = mh & 3;
  int rowBase = blockIdx.x * 64 + wave * 16;
  const half_t* Wt; const float* bias; half_t* out; int vmode = 0;
  if (blockIdx.z == 0)      { Wt = Wqt; bias = bq; out = Q; }
  else if (blockIdx.z == 1) { Wt = Wkt; bias = bk; out = K; }
  else                      { Wt = Wvt; bias = bv; out = V; vmode = 1; }
  const half_t* A  = X + (size_t)m * modStride + (size_t)rowBase * lda;
  const half_t* Wb = Wt + (size_t)mh * din * HD;   // [64][din] f16
  v8f acc[4] = {};
  for (int k0 = 0; k0 < din; k0 += 32) {
    v16h a = load_frag16(A + k0, lda, lane);
#pragma unroll
    for (int f = 0; f < 4; ++f) {
      v16h bfr = load_frag16(Wb + (size_t)(f * 16) * din + k0, din, lane);
      acc[f] = wmma16(a, bfr, acc[f]);
    }
  }
#pragma unroll
  for (int f = 0; f < 4; ++f)
#pragma unroll
    for (int e = 0; e < 8; ++e) {
      int row = rowBase + e + ((lane >> 4) << 3);   // row in BS
      int n   = f * 16 + (lane & 15);
      int bb  = row >> 9, s = row & (SS - 1);
      size_t headOff = ((((size_t)m * BB + bb) * 4 + h) * SS) * HD;
      float v = acc[f][e] + bias[mh * HD + n];
      if (vmode) out[headOff + (size_t)n * SS + s] = (half_t)v;
      else       out[headOff + (size_t)s * HD + n] = (half_t)v;
    }
}

// ---------------- fused attention per (m,b,h) ----------------
// grid = (S/64, 192). 4 waves; wave w owns 16 query rows. 128 KB LDS.
__global__ __launch_bounds__(128) void attn_kernel(
    const half_t* __restrict__ Q, const half_t* __restrict__ K,
    const half_t* __restrict__ V, half_t* __restrict__ ctx) {
  __shared__ __align__(16) half_t sc[4][16][SS];   // scores, 64 KB
  __shared__ __align__(16) half_t pr[4][16][SS];   // probs,  64 KB
  int wave = threadIdx.x >> 5, lane = threadIdx.x & 31;
  int y = blockIdx.y;
  int m = y >> 6, bb = (y >> 2) & 15, h = y & 3;
  int rowBase = blockIdx.x * 64 + wave * 16;
  size_t headOff = (size_t)y * SS * HD;
  const half_t* Qh = Q + headOff + (size_t)rowBase * HD;
  const half_t* Kh = K + headOff;         // [S][64]
  const half_t* Vh = V + headOff;         // [64][S]  (transposed)

  v16h aq0 = load_frag16(Qh,      HD, lane);   // K = 0..31
  v16h aq1 = load_frag16(Qh + 32, HD, lane);   // K = 32..63

  for (int j = 0; j < SS / 16; ++j) {
    const half_t* Kt = Kh + (size_t)j * 16 * HD;
    if (j + 1 < SS / 16) __builtin_prefetch(Kt + 16 * HD, 0, 1);
    v16h b0 = load_frag16(Kt,      HD, lane);
    v16h b1 = load_frag16(Kt + 32, HD, lane);
    v8f z = {};
    v8f s = wmma16(aq1, b1, wmma16(aq0, b0, z));
#pragma unroll
    for (int e = 0; e < 8; ++e) {
      int r = e + ((lane >> 4) << 3);
      sc[wave][r][j * 16 + (lane & 15)] = (half_t)(s[e] * 0.125f);  // 1/sqrt(64)
    }
  }
  __syncthreads();

  // softmax: 2 lanes per row, 256 elements each
  {
    int r = lane >> 1, hf = lane & 1;
    const half_t* srow = &sc[wave][r][hf * 256];
    half_t* prow = &pr[wave][r][hf * 256];
    float mx = -1e30f;
    for (int i = 0; i < 256; ++i) mx = fmaxf(mx, (float)srow[i]);
    mx = fmaxf(mx, __shfl_xor(mx, 1, 32));
    float sum = 0.f;
    for (int i = 0; i < 256; ++i) {
      float p = expf((float)srow[i] - mx);
      prow[i] = (half_t)p;
      sum += p;
    }
    sum += __shfl_xor(sum, 1, 32);
    float inv = 1.f / sum;
    for (int i = 0; i < 256; ++i) prow[i] = (half_t)((float)prow[i] * inv);
  }
  __syncthreads();

  // context = probs (16x512, LDS) x V (512x64, via V^T)
  v8f co[4] = {};
  for (int j = 0; j < SS / 32; ++j) {
    v16h a = load_frag16(&pr[wave][0][j * 32], SS, lane);   // ds_load_b128 frags
#pragma unroll
    for (int f = 0; f < 4; ++f) {
      v16h bfr = load_frag16(Vh + (size_t)(f * 16) * SS + j * 32, SS, lane);
      co[f] = wmma16(a, bfr, co[f]);
    }
  }
#pragma unroll
  for (int f = 0; f < 4; ++f)
#pragma unroll
    for (int e = 0; e < 8; ++e) {
      int s   = rowBase + e + ((lane >> 4) << 3);
      int n   = f * 16 + (lane & 15);
      int row = bb * SS + s;
      int col = m * 256 + h * HD + n;
      ctx[(size_t)row * HH + col] = (half_t)co[f][e];
    }
}

// ---------------- generic block-diagonal GEMM ----------------
// out = act( A[row, nb*aColPerNb + k] * W_nb[k,n] + bias (+ R) )
// Wt: transposed f16 weights [nb][N][K]. grid = (BS/64, 3*tilesPerNb).
__global__ __launch_bounds__(128) void blk_gemm(
    const half_t* __restrict__ A, int lda, int aColPerNb,
    const half_t* __restrict__ Wt, long wPerNb,
    const float* __restrict__ bias, int biasPerNb,
    const half_t* __restrict__ R, int ldr, int rColPerNb,
    float* __restrict__ Cf, half_t* __restrict__ Ch, int ldc, int cColPerNb,
    int Kd, int tilesPerNb, int doGelu) {
  int wave = threadIdx.x >> 5, lane = threadIdx.x & 31;
  int nb = blockIdx.y / tilesPerNb, nt = blockIdx.y % tilesPerNb;
  int rowBase = blockIdx.x * 64 + wave * 16;
  const half_t* Ab = A + (size_t)rowBase * lda + nb * aColPerNb;
  const half_t* Wb = Wt + (size_t)nb * wPerNb + (size_t)(nt * 64) * Kd;
  v8f acc[4] = {};
  for (int k0 = 0; k0 < Kd; k0 += 32) {
    if (k0 + 32 < Kd) __builtin_prefetch(Ab + k0 + 32, 0, 1);
    v16h a = load_frag16(Ab + k0, lda, lane);
#pragma unroll
    for (int f = 0; f < 4; ++f) {
      v16h bfr = load_frag16(Wb + (size_t)(f * 16) * Kd + k0, Kd, lane);
      acc[f] = wmma16(a, bfr, acc[f]);
    }
  }
#pragma unroll
  for (int f = 0; f < 4; ++f)
#pragma unroll
    for (int e = 0; e < 8; ++e) {
      int row = rowBase + e + ((lane >> 4) << 3);
      int n   = nt * 64 + f * 16 + (lane & 15);
      float v = acc[f][e] + bias[nb * biasPerNb + n];
      if (R) v += (float)R[(size_t)row * ldr + nb * rColPerNb + n];
      if (doGelu) v = 0.5f * v * (1.f + erff(v * 0.70710678118654752f));
      size_t off = (size_t)row * ldc + nb * cColPerNb + n;
      if (Cf) Cf[off] = v;
      else    Ch[off] = (half_t)v;
    }
}

// ---------------- block LayerNorm over 256-wide groups ----------------
// in: f32; writes f32 (outf) or f16 (outh), whichever is non-null.
__global__ __launch_bounds__(128) void block_ln(
    const float* __restrict__ in, const float* __restrict__ w,
    const float* __restrict__ b, float* __restrict__ outf,
    half_t* __restrict__ outh, int rows) {
  int gid  = blockIdx.x * 4 + (threadIdx.x >> 5);
  int lane = threadIdx.x & 31;
  if (gid >= rows * 3) return;
  int row = gid / 3, nb = gid % 3;
  const float* p = in + (size_t)row * HH + nb * 256;
  float vals[8];
  float sum = 0.f;
#pragma unroll
  for (int i = 0; i < 8; ++i) { vals[i] = p[lane + i * 32]; sum += vals[i]; }
#pragma unroll
  for (int off = 16; off >= 1; off >>= 1) sum += __shfl_xor(sum, off, 32);
  float mean = sum * (1.f / 256.f);
  float var = 0.f;
#pragma unroll
  for (int i = 0; i < 8; ++i) { float d = vals[i] - mean; var += d * d; }
#pragma unroll
  for (int off = 16; off >= 1; off >>= 1) var += __shfl_xor(var, off, 32);
  float inv = rsqrtf(var * (1.f / 256.f) + EPS);
#pragma unroll
  for (int i = 0; i < 8; ++i) {
    int c = lane + i * 32;
    float v = (vals[i] - mean) * inv * w[nb * 256 + c] + b[nb * 256 + c];
    size_t off = (size_t)row * HH + nb * 256 + c;
    if (outf) outf[off] = v;
    else      outh[off] = (half_t)v;
  }
}

// ---------------- host-side orchestration ----------------
extern "C" void kernel_launch(void* const* d_in, const int* in_sizes, int n_in,
                              void* d_out, int out_size, void* d_ws, size_t ws_size,
                              hipStream_t stream) {
  (void)in_sizes; (void)n_in; (void)out_size; (void)ws_size;
  const float* e1   = (const float*)d_in[0];
  const float* e2   = (const float*)d_in[1];
  const float* e3   = (const float*)d_in[2];
  const float* pos  = (const float*)d_in[3];
  const float* elnw = (const float*)d_in[4];
  const float* elnb = (const float*)d_in[5];

  struct LayerP {
    const float *Wq,*bq,*Wk,*bk,*Wv,*bv;
    const float *soW,*sob,*solnw,*solnb,*intW,*intb,*outW,*outb,*olnw,*olnb;
    int din;
  };
  LayerP L[4];
  for (int i = 0; i < 4; ++i) {
    int base = 6 + i * 16;
    L[i].Wq    = (const float*)d_in[base + 0];
    L[i].bq    = (const float*)d_in[base + 1];
    L[i].Wk    = (const float*)d_in[base + 2];
    L[i].bk    = (const float*)d_in[base + 3];
    L[i].Wv    = (const float*)d_in[base + 4];
    L[i].bv    = (const float*)d_in[base + 5];
    L[i].soW   = (const float*)d_in[base + 6];
    L[i].sob   = (const float*)d_in[base + 7];
    L[i].solnw = (const float*)d_in[base + 8];
    L[i].solnb = (const float*)d_in[base + 9];
    L[i].intW  = (const float*)d_in[base + 10];
    L[i].intb  = (const float*)d_in[base + 11];
    L[i].outW  = (const float*)d_in[base + 12];
    L[i].outb  = (const float*)d_in[base + 13];
    L[i].olnw  = (const float*)d_in[base + 14];
    L[i].olnb  = (const float*)d_in[base + 15];
    L[i].din   = (i == 0) ? HH : (HH / 3);
  }

  const size_t N1 = (size_t)BS * HH;     // 6,291,456 elements
  half_t* hws  = (half_t*)d_ws;
  half_t* o16   = hws;                   // [3][BS][768]
  half_t* res16 = o16 + 3 * N1;
  half_t* Q16   = res16 + N1;            // [(m,b,h)][S][64]
  half_t* K16   = Q16 + N1;
  half_t* V16   = K16 + N1;              // [(m,b,h)][64][S] (transposed)
  half_t* ctx16 = V16 + N1;
  half_t* h16   = ctx16 + N1;
  half_t* x16   = h16 + N1;
  half_t* int16b= x16 + N1;              // [BS][3072] = 4*N1
  half_t* wp    = int16b + 4 * N1;       // f16 weight region

  struct LayerW { half_t *Wqt,*Wkt,*Wvt,*soWt,*intWt,*outWt; };
  LayerW W[4];
  for (int i = 0; i < 4; ++i) {
    size_t qkvSz = (size_t)12 * L[i].din * HD;
    W[i].Wqt  = wp; wp += qkvSz;
    W[i].Wkt  = wp; wp += qkvSz;
    W[i].Wvt  = wp; wp += qkvSz;
    W[i].soWt = wp; wp += (size_t)3 * 256 * 256;
    W[i].intWt= wp; wp += (size_t)3 * 256 * 1024;
    W[i].outWt= wp; wp += (size_t)3 * 1024 * 256;
  }
  float* pre = (float*)wp;               // f32 pre-LN buffer, N1 floats

  // --- convert + transpose all weights to f16 [N][K] ---
  for (int i = 0; i < 4; ++i) {
    int din = L[i].din;
    int qg = (din * HD + 255) / 256;
    wt_cvt<<<dim3(qg, 12), 256, 0, stream>>>(L[i].Wq, W[i].Wqt, din, HD);
    wt_cvt<<<dim3(qg, 12), 256, 0, stream>>>(L[i].Wk, W[i].Wkt, din, HD);
    wt_cvt<<<dim3(qg, 12), 256, 0, stream>>>(L[i].Wv, W[i].Wvt, din, HD);
    wt_cvt<<<dim3((256 * 256 + 255) / 256, 3), 256, 0, stream>>>(L[i].soW,  W[i].soWt,  256, 256);
    wt_cvt<<<dim3((256 * 1024 + 255) / 256, 3), 256, 0, stream>>>(L[i].intW, W[i].intWt, 256, 1024);
    wt_cvt<<<dim3((1024 * 256 + 255) / 256, 3), 256, 0, stream>>>(L[i].outW, W[i].outWt, 1024, 256);
  }

  embed_ln<<<(3 * BS) / 4, 128, 0, stream>>>(e1, e2, e3, pos, elnw, elnb, o16);
  res_avg<<<(int)((N1 + 255) / 256), 256, 0, stream>>>(o16, res16);

  for (int l = 0; l < 4; ++l) {
    const LayerP& p = L[l];
    const half_t* X; long modStride;
    if (l == 0) { X = o16; modStride = (long)N1; }   // per-modality full hidden
    else        { X = x16; modStride = 256; }        // third-slice of fused hidden

    qkv_gemm<<<dim3(BS / 64, 12, 3), 128, 0, stream>>>(
        X, modStride, HH, p.din, W[l].Wqt, p.bq, W[l].Wkt, p.bk, W[l].Wvt, p.bv,
        Q16, K16, V16);

    attn_kernel<<<dim3(SS / 64, 192), 128, 0, stream>>>(Q16, K16, V16, ctx16);

    const half_t* R = (l == 0) ? res16 : x16;
    blk_gemm<<<dim3(BS / 64, 12), 128, 0, stream>>>(
        ctx16, HH, 256, W[l].soWt, 256L * 256, p.sob, 256,
        R, HH, 256, pre, (half_t*)nullptr, HH, 256, 256, 4, 0);
    block_ln<<<(BS * 3) / 4, 128, 0, stream>>>(pre, p.solnw, p.solnb,
                                               (float*)nullptr, h16, BS);

    blk_gemm<<<dim3(BS / 64, 48), 128, 0, stream>>>(
        h16, HH, 256, W[l].intWt, 256L * 1024, p.intb, 1024,
        (const half_t*)nullptr, 0, 0, (float*)nullptr, int16b, 3072, 1024,
        256, 16, 1);

    blk_gemm<<<dim3(BS / 64, 12), 128, 0, stream>>>(
        int16b, 3072, 1024, W[l].outWt, 1024L * 256, p.outb, 256,
        h16, HH, 256, pre, (half_t*)nullptr, HH, 256, 1024, 4, 0);

    if (l == 3)
      block_ln<<<(BS * 3) / 4, 128, 0, stream>>>(pre, p.olnw, p.olnb,
                                                 (float*)d_out, (half_t*)nullptr, BS);
    else
      block_ln<<<(BS * 3) / 4, 128, 0, stream>>>(pre, p.olnw, p.olnb,
                                                 (float*)nullptr, x16, BS);
  }
}
